// CausalSelfAttention_18519898980516
// MI455X (gfx1250) — matrix operand order repaired and checked by reading, not compiled, not run on hardware
//
#include <hip/hip_runtime.h>
#include <math.h>

// ---------------------------------------------------------------------------
// Causal self-attention with RoPE for MI455X (gfx1250, wave32, WMMA).
// fp32 inputs -> bf16 compute on v_wmma_f32_16x16x32_bf16 -> fp32 output.
// ---------------------------------------------------------------------------

typedef __bf16 bf16;
typedef __attribute__((ext_vector_type(16))) __bf16 v16bf;
typedef __attribute__((ext_vector_type(8)))  __bf16 v8bf;
typedef __attribute__((ext_vector_type(8)))  float  v8f;

constexpr int BB  = 2;     // batch
constexpr int SS  = 2048;  // sequence
constexpr int DD  = 4096;  // model dim
constexpr int HH  = 32;    // heads
constexpr int HD  = 128;   // head dim
constexpr int MR  = BB * SS;                 // 4096 rows
constexpr size_t MAT = (size_t)DD * DD;      // 16777216 elems; every ws buffer

__device__ __forceinline__ v16bf pack16(v8bf lo, v8bf hi) {
  v16bf r;
#pragma unroll
  for (int e = 0; e < 8; ++e) { r[e] = lo[e]; r[e + 8] = hi[e]; }
  return r;
}

// ---------------------------------------------------------------------------
// fp32 -> bf16 conversion (bandwidth trivial vs. the GEMMs)
// ---------------------------------------------------------------------------
__global__ void cvt_f32_bf16(const float* __restrict__ src,
                             bf16* __restrict__ dst, int n) {
  for (int i = blockIdx.x * blockDim.x + threadIdx.x; i < n;
       i += gridDim.x * blockDim.x)
    dst[i] = (bf16)src[i];
}

// ---------------------------------------------------------------------------
// GEMM  C[m,n] = sum_k A[m,k] * W[n,k]     (i.e. y = x @ W^T)
// A:[M,K] bf16 row-major, W:[N,K] bf16 row-major (both read along contiguous K).
// Block = 4 waves, 128x128 tile, LDS-staged K=32 panels (8KB A + 8KB B).
// Each wave computes 64x64 = 16 WMMAs per K-step from LDS fragments.
// LDS rows padded to 40 elems (80B) -> the 16 row-parallel b128 fragment
// reads land on disjoint bank groups (r*20 mod 64 has full period 16).
// MODE 0: fp32 row-major [M,N]                  (final output projection)
// MODE 1: bf16 scattered to [B,H,S,HD]          (Q, K)
// MODE 2: bf16 scattered to [B,H,HD,S]          (V^T for the PV B-fragments)
// ---------------------------------------------------------------------------
template <int MODE>
__global__ __launch_bounds__(128)
void gemm_bf16_nt(const bf16* __restrict__ A, const bf16* __restrict__ W,
                  void* __restrict__ out, int M, int N, int K) {
  constexpr int LSTR = 40;  // padded LDS row stride (elements)
  __shared__ __align__(16) bf16 As[128 * LSTR];
  __shared__ __align__(16) bf16 Bs[128 * LSTR];

  const int tid  = threadIdx.x;
  const int lane = tid & 31;
  const int widx = tid >> 5;
  const int r    = lane & 15;   // tile column / fragment row
  const int hi   = lane >> 4;   // half-wave select
  const int wm   = (widx >> 1) * 64;  // wave M offset inside block tile
  const int wn   = (widx & 1) * 64;   // wave N offset inside block tile
  const int m0   = blockIdx.y * 128;
  const int n0   = blockIdx.x * 128;

  v8f zero = {};
  v8f acc[4][4];
#pragma unroll
  for (int mt = 0; mt < 4; ++mt)
#pragma unroll
    for (int nt = 0; nt < 4; ++nt) acc[mt][nt] = zero;

  for (int k0 = 0; k0 < K; k0 += 32) {
    // ---- cooperative global -> LDS staging: one 32-elem row per thread ----
    const v8bf* ga = (const v8bf*)(A + (size_t)(m0 + tid) * K + k0);
    const v8bf* gb = (const v8bf*)(W + (size_t)(n0 + tid) * K + k0);
    v8bf a0 = ga[0], a1 = ga[1], a2 = ga[2], a3 = ga[3];
    v8bf b0 = gb[0], b1 = gb[1], b2 = gb[2], b3 = gb[3];
    if (k0 + 32 < K) {  // hint next K panel (global_prefetch_b8)
      __builtin_prefetch(A + (size_t)(m0 + tid) * K + k0 + 32, 0, 0);
      __builtin_prefetch(W + (size_t)(n0 + tid) * K + k0 + 32, 0, 0);
    }
    __syncthreads();  // previous iteration's LDS reads done
    *(v8bf*)&As[tid * LSTR + 0]  = a0;
    *(v8bf*)&As[tid * LSTR + 8]  = a1;
    *(v8bf*)&As[tid * LSTR + 16] = a2;
    *(v8bf*)&As[tid * LSTR + 24] = a3;
    *(v8bf*)&Bs[tid * LSTR + 0]  = b0;
    *(v8bf*)&Bs[tid * LSTR + 8]  = b1;
    *(v8bf*)&Bs[tid * LSTR + 16] = b2;
    *(v8bf*)&Bs[tid * LSTR + 24] = b3;
    __syncthreads();  // panel visible to all waves

    // ---- fragments from LDS ----
    v16bf aF[4];
#pragma unroll
    for (int mt = 0; mt < 4; ++mt) {
      // A fragment 16x32: lane holds row m, K = hi*16 + [0..15]
      const bf16* p = &As[(wm + mt * 16 + r) * LSTR];
      aF[mt] = pack16(*(const v8bf*)(p + hi * 16),
                      *(const v8bf*)(p + hi * 16 + 8));
    }
    v16bf bF[4];
#pragma unroll
    for (int nt = 0; nt < 4; ++nt) {
      // B fragment 32x16: lane holds col n (= W row n), K = hi*16 + [0..15]
      const bf16* p = &Bs[(wn + nt * 16 + r) * LSTR + hi * 16];
      bF[nt] = pack16(*(const v8bf*)p, *(const v8bf*)(p + 8));
    }
#pragma unroll
    for (int mt = 0; mt < 4; ++mt)
#pragma unroll
      for (int nt = 0; nt < 4; ++nt)
        acc[mt][nt] = __builtin_amdgcn_wmma_f32_16x16x32_bf16(
            false, aF[mt], false, bF[nt], (short)0, acc[mt][nt], false, false);
  }

  // C/D layout: VGPR i -> row M = hi*8 + i, lane -> col N = r
#pragma unroll
  for (int mt = 0; mt < 4; ++mt)
#pragma unroll
    for (int nt = 0; nt < 4; ++nt)
#pragma unroll
      for (int i = 0; i < 8; ++i) {
        const int m = m0 + wm + mt * 16 + hi * 8 + i;
        const int n = n0 + wn + nt * 16 + r;
        const float v = acc[mt][nt][i];
        if (MODE == 0) {
          ((float*)out)[(size_t)m * N + n] = v;
        } else {
          const int b = m >> 11, s = m & (SS - 1);   // m = b*S + s
          const int h = n >> 7,  hd = n & (HD - 1);  // n = h*HD + hd
          if (MODE == 1)
            ((bf16*)out)[((size_t)(b * HH + h) * SS + s) * HD + hd] = (bf16)v;
          else
            ((bf16*)out)[((size_t)(b * HH + h) * HD + hd) * SS + s] = (bf16)v;
        }
      }
}

// ---------------------------------------------------------------------------
// RoPE in-place on [B,H,S,HD] bf16; each thread handles the (j, j+64) pair.
// scale folds 1/sqrt(HD) into Q.
// ---------------------------------------------------------------------------
__global__ void rope_kernel(bf16* __restrict__ t, float scale, int total) {
  for (int idx = blockIdx.x * blockDim.x + threadIdx.x; idx < total;
       idx += gridDim.x * blockDim.x) {
    const int j = idx & 63;
    const int s = (idx >> 6) & (SS - 1);
    const size_t base = (size_t)(idx >> 6) * HD;  // ((b*H+h)*S + s) * HD
    // inv_freq[j] = 10000^(-j/64) = exp(-j * ln(1e4)/64)
    const float ang = (float)s * __expf(-(float)j * 0.14391156f);
    float c, si;
    __sincosf(ang, &si, &c);
    const float x1 = (float)t[base + j];
    const float x2 = (float)t[base + 64 + j];
    t[base + j]      = (bf16)((x1 * c - x2 * si) * scale);
    t[base + 64 + j] = (bf16)((x2 * c + x1 * si) * scale);
  }
}

// ---------------------------------------------------------------------------
// Flash attention: one wave per 16-row Q tile of one (b,h). Keys streamed
// 32 at a time: QK^T = 8 wmma / step, online softmax in C-fragment layout,
// P goes through LDS to be re-read in A-fragment layout, PV = 8 wmma / step
// against V^T [B,H,HD,S]. Scale already folded into Q.
// ---------------------------------------------------------------------------
__global__ __launch_bounds__(128)
void attn_kernel(const bf16* __restrict__ q, const bf16* __restrict__ k,
                 const bf16* __restrict__ vt, bf16* __restrict__ ctx) {
  __shared__ __align__(16) bf16 Pt[4][16 * 32];  // per-wave P tile (1KB)
  const int lane = threadIdx.x & 31;
  const int widx = threadIdx.x >> 5;
  const int r  = lane & 15;
  const int hi = lane >> 4;
  const int gw = blockIdx.x * 4 + widx;        // global wave id
  const int bh = gw / (SS / 16);               // (b*H + h)
  const int q0 = (gw % (SS / 16)) * 16;        // first query row of tile

  const bf16* qp = q  + (size_t)bh * SS * HD;
  const bf16* kp = k  + (size_t)bh * SS * HD;
  const bf16* vp = vt + (size_t)bh * HD * SS;

  // Q A-fragments for all 4 K-chunks of HD=128 (kept resident).
  v16bf aQ[4];
#pragma unroll
  for (int c = 0; c < 4; ++c) {
    const bf16* p = qp + (size_t)(q0 + r) * HD + 32 * c;
    aQ[c] = pack16(*(const v8bf*)(p + hi * 16),
                   *(const v8bf*)(p + hi * 16 + 8));
  }

  v8f zero = {};
  v8f acc[8];                  // O[i, d] : 8 d-tiles of 16, C layout
  float m_i[8], l_i[8];
#pragma unroll
  for (int i = 0; i < 8; ++i) { m_i[i] = -1e30f; l_i[i] = 0.f; }
#pragma unroll
  for (int n = 0; n < 8; ++n) acc[n] = zero;

  for (int j0 = 0; j0 < q0 + 16; j0 += 32) {
    // --- scores: two 16x16 tiles (keys j0..15, j0+16..31)
    v8f sc[2] = {zero, zero};
#pragma unroll
    for (int jt = 0; jt < 2; ++jt)
#pragma unroll
      for (int c = 0; c < 4; ++c) {
        const bf16* p = kp + (size_t)(j0 + jt * 16 + r) * HD + 32 * c + hi * 16;
        v16bf bK = pack16(*(const v8bf*)p, *(const v8bf*)(p + 8));
        sc[jt] = __builtin_amdgcn_wmma_f32_16x16x32_bf16(
            false, aQ[c], false, bK, (short)0, sc[jt], false, false);
      }

    // --- causal mask on diagonal blocks (row = q0+hi*8+i, col = j0+jt*16+r)
    if (j0 + 31 > q0) {
#pragma unroll
      for (int jt = 0; jt < 2; ++jt)
#pragma unroll
        for (int i = 0; i < 8; ++i)
          if (j0 + jt * 16 + r > q0 + hi * 8 + i) sc[jt][i] = -1e30f;
    }

    // --- row max across the 16 column-lanes of each half-wave
    float mn[8];
#pragma unroll
    for (int i = 0; i < 8; ++i) mn[i] = fmaxf(sc[0][i], sc[1][i]);
#pragma unroll
    for (int off = 1; off < 16; off <<= 1)
#pragma unroll
      for (int i = 0; i < 8; ++i)
        mn[i] = fmaxf(mn[i], __shfl_xor(mn[i], off, 16));

    // --- online softmax update
    float rs[8];
#pragma unroll
    for (int i = 0; i < 8; ++i) {
      const float mt = fmaxf(m_i[i], mn[i]);
      const float al = __expf(m_i[i] - mt);
      m_i[i] = mt;
      l_i[i] *= al;
#pragma unroll
      for (int n = 0; n < 8; ++n) acc[n][i] *= al;
      const float p0 = __expf(sc[0][i] - mt);
      const float p1 = __expf(sc[1][i] - mt);
      sc[0][i] = p0; sc[1][i] = p1;
      rs[i] = p0 + p1;
    }
#pragma unroll
    for (int off = 1; off < 16; off <<= 1)
#pragma unroll
      for (int i = 0; i < 8; ++i) rs[i] += __shfl_xor(rs[i], off, 16);
#pragma unroll
    for (int i = 0; i < 8; ++i) l_i[i] += rs[i];

    // --- P (C layout) -> LDS -> A-fragment layout
#pragma unroll
    for (int jt = 0; jt < 2; ++jt)
#pragma unroll
      for (int i = 0; i < 8; ++i)
        Pt[widx][(hi * 8 + i) * 32 + jt * 16 + r] = (bf16)sc[jt][i];
    __asm__ volatile("s_wait_dscnt 0" ::: "memory");
    v16bf aP = pack16(*(const v8bf*)&Pt[widx][r * 32 + hi * 16],
                      *(const v8bf*)&Pt[widx][r * 32 + hi * 16 + 8]);

    // --- PV: B fragment = 16 contiguous keys from a row of V^T
#pragma unroll
    for (int n = 0; n < 8; ++n) {
      const bf16* p = vp + (size_t)(n * 16 + r) * SS + j0 + hi * 16;
      v16bf bV = pack16(*(const v8bf*)p, *(const v8bf*)(p + 8));
      acc[n] = __builtin_amdgcn_wmma_f32_16x16x32_bf16(
          false, aP, false, bV, (short)0, acc[n], false, false);
    }
  }

  // --- epilogue: ctx[b, s, h*HD + d] bf16, normalized by row sum
  const int b = bh / HH, h = bh % HH;
#pragma unroll
  for (int n = 0; n < 8; ++n)
#pragma unroll
    for (int i = 0; i < 8; ++i) {
      const int row = q0 + hi * 8 + i;
      ctx[((size_t)(b * SS + row)) * DD + h * HD + n * 16 + r] =
          (bf16)(acc[n][i] / l_i[i]);
    }
}

// ---------------------------------------------------------------------------
// Launch: 5 converts -> 3 projection GEMMs -> 2 RoPE -> attention -> out GEMM
// Workspace: 9 bf16 buffers x 16.78M elems = 288 MB.
// ---------------------------------------------------------------------------
extern "C" void kernel_launch(void* const* d_in, const int* in_sizes, int n_in,
                              void* d_out, int out_size, void* d_ws,
                              size_t ws_size, hipStream_t stream) {
  (void)in_sizes; (void)n_in; (void)out_size; (void)ws_size;
  const float* x  = (const float*)d_in[0];
  const float* Wq = (const float*)d_in[1];
  const float* Wk = (const float*)d_in[2];
  const float* Wv = (const float*)d_in[3];
  const float* Wo = (const float*)d_in[4];
  float* out = (float*)d_out;

  bf16* xb  = (bf16*)d_ws;
  bf16* wqb = xb  + MAT;
  bf16* wkb = wqb + MAT;
  bf16* wvb = wkb + MAT;
  bf16* wob = wvb + MAT;
  bf16* qb  = wob + MAT;
  bf16* kb  = qb  + MAT;
  bf16* vtb = kb  + MAT;
  bf16* cxb = vtb + MAT;

  const int n = (int)MAT;
  cvt_f32_bf16<<<4096, 256, 0, stream>>>(x,  xb,  n);
  cvt_f32_bf16<<<4096, 256, 0, stream>>>(Wq, wqb, n);
  cvt_f32_bf16<<<4096, 256, 0, stream>>>(Wk, wkb, n);
  cvt_f32_bf16<<<4096, 256, 0, stream>>>(Wv, wvb, n);
  cvt_f32_bf16<<<4096, 256, 0, stream>>>(Wo, wob, n);

  dim3 gb(128);
  dim3 gg(DD / 128, MR / 128);
  gemm_bf16_nt<1><<<gg, gb, 0, stream>>>(xb, wqb, qb,  MR, DD, DD);
  gemm_bf16_nt<1><<<gg, gb, 0, stream>>>(xb, wkb, kb,  MR, DD, DD);
  gemm_bf16_nt<2><<<gg, gb, 0, stream>>>(xb, wvb, vtb, MR, DD, DD);

  const int pairs = BB * HH * SS * (HD / 2);
  rope_kernel<<<4096, 256, 0, stream>>>(qb, 0.08838834764831845f, pairs); // 1/sqrt(128)
  rope_kernel<<<4096, 256, 0, stream>>>(kb, 1.0f, pairs);

  attn_kernel<<<(BB * HH * (SS / 16)) / 4, 128, 0, stream>>>(qb, kb, vtb, cxb);

  gemm_bf16_nt<0><<<gg, gb, 0, stream>>>(cxb, wob, out, MR, DD, DD);
}
